// Attention_57432302682403
// MI455X (gfx1250) — matrix operand order repaired and checked by reading, not compiled, not run on hardware
//
#include <hip/hip_runtime.h>
#include <math.h>

// Shapes fixed by the reference: b=2, n=2048, dim=1024, h=16, dh=64
#define BATCH     2
#define SEQ       2048
#define DMODEL    1024
#define NHEADS    16
#define DHEAD     64

typedef __attribute__((ext_vector_type(16))) _Float16 v16h;
typedef __attribute__((ext_vector_type(8)))  float    v8f;

// ---------------------------------------------------------------------------
// WMMA fragment helpers (layouts per CDNA5 ISA 7.12.2, wave32)
// ---------------------------------------------------------------------------

__device__ __forceinline__ v8f v8f_zero() {
    v8f z;
#pragma unroll
    for (int i = 0; i < 8; ++i) z[i] = 0.0f;
    return z;
}

// A-matrix 16x32 f16, row-major source with leading dim lda (elements).
// lane L: m = m0 + L%16, g = L/16; elems[0..7] = K = k0+8g .. k0+8g+7,
// elems[8..15] = K = k0+16+8g .. +7   (VGPR0..3: K pairs 8g..; VGPR4..7: 16+8g..)
__device__ __forceinline__ v16h load_a16(const _Float16* base, int lda,
                                         int m0, int k0, int lane) {
    const int m = m0 + (lane & 15);
    const int g = (lane >> 4) & 1;
    const _Float16* p = base + (size_t)m * lda + k0 + (g << 3);
    v16h r;
#pragma unroll
    for (int i = 0; i < 8; ++i) { r[i] = p[i]; r[i + 8] = p[i + 16]; }
    return r;
}

// B-matrix 32x16 f16 where B[k][n] = W[n][k], W row-major leading dim ldw.
// lane L: n = n0 + L%16, g = L/16; elems[0..7] = K = k0+8g .. +7, elems[8..15] = K = k0+16+8g .. +7.
__device__ __forceinline__ v16h load_b16(const _Float16* W, int ldw,
                                         int n0, int k0, int lane) {
    const int n = n0 + (lane & 15);
    const int g = (lane >> 4) & 1;
    const _Float16* p = W + (size_t)n * ldw + k0 + (g << 3);
    v16h r;
#pragma unroll
    for (int i = 0; i < 8; ++i) { r[i] = p[i]; r[i + 8] = p[i + 16]; }
    return r;
}

__device__ __forceinline__ v8f wmma_f16(v16h a, v16h b, v8f c) {
    return __builtin_amdgcn_wmma_f32_16x16x32_f16(
        /*neg_a=*/false, a, /*neg_b=*/false, b,
        /*c_mod=*/(short)0, c, /*reuse_a=*/false, /*reuse_b=*/false);
}

__device__ __forceinline__ void wait_ds0() {
#if __has_builtin(__builtin_amdgcn_s_wait_dscnt)
    __builtin_amdgcn_s_wait_dscnt(0);
#else
    asm volatile("s_wait_dscnt 0" ::: "memory");
#endif
}

// ---------------------------------------------------------------------------
// Kernel 1: fp32 -> f16 convert
// ---------------------------------------------------------------------------
__global__ void k_convert_f16(const float* __restrict__ src,
                              _Float16* __restrict__ dst, int n) {
    int i = blockIdx.x * blockDim.x + threadIdx.x;
    if (i < n) dst[i] = (_Float16)src[i];
}

// ---------------------------------------------------------------------------
// Kernel 2: generic WMMA GEMM  Y[m,n] = sum_k A[m,k] * W[n,k]
//   mode 0: write f16 to head-major layout (b, h, seq, 64)
//   mode 1: write fp32 row-major + bias (final projection into d_out)
// Block: 256 threads = 8 waves (2x4); wave tile 32x64; block tile 64x256.
// Per k-step: 2 A frags + 4 B frags -> 8 WMMAs (A reused 4x, B reused 2x).
// ---------------------------------------------------------------------------
__global__ __launch_bounds__(256)
void k_gemm_wmma(const _Float16* __restrict__ A, const _Float16* __restrict__ W,
                 int M, int N, int K, int seqlen,
                 _Float16* __restrict__ outH, float* __restrict__ outF,
                 const float* __restrict__ bias, int mode) {
    const int lane = threadIdx.x & 31;
    const int wave = threadIdx.x >> 5;
    const int m0 = blockIdx.x * 64 + (wave >> 2) * 32;     // 2 wave rows
    const int n0 = blockIdx.y * 256 + (wave & 3) * 64;     // 4 wave cols

    v8f acc[2][4];
#pragma unroll
    for (int sm = 0; sm < 2; ++sm)
#pragma unroll
        for (int sn = 0; sn < 4; ++sn) acc[sm][sn] = v8f_zero();

    for (int k0 = 0; k0 < K; k0 += 32) {
        // prefetch next k-tile toward the WGP while this step computes
        if (k0 + 32 < K) {
            __builtin_prefetch(A + (size_t)(m0 + (lane & 15)) * K + k0 + 32, 0, 3);
            __builtin_prefetch(W + (size_t)(n0 + (lane & 15)) * K + k0 + 32, 0, 3);
            __builtin_prefetch(W + (size_t)(n0 + 32 + (lane & 15)) * K + k0 + 32, 0, 3);
        }
        v16h a0 = load_a16(A, K, m0, k0, lane);
        v16h a1 = load_a16(A, K, m0 + 16, k0, lane);
#pragma unroll
        for (int sn = 0; sn < 4; ++sn) {
            v16h b = load_b16(W, K, n0 + sn * 16, k0, lane);
            acc[0][sn] = wmma_f16(a0, b, acc[0][sn]);
            acc[1][sn] = wmma_f16(a1, b, acc[1][sn]);
        }
    }

    const int hv = lane >> 4, nl = lane & 15;
#pragma unroll
    for (int sm = 0; sm < 2; ++sm) {
#pragma unroll
        for (int sn = 0; sn < 4; ++sn) {
#pragma unroll
            for (int i = 0; i < 8; ++i) {
                const int row = m0 + sm * 16 + i + 8 * hv;   // C layout: M = vgpr + 8*half
                const int col = n0 + sn * 16 + nl;           //           N = lane%16
                const float v = acc[sm][sn][i];
                if (mode == 0) {
                    const int bidx = row / seqlen, nn = row % seqlen;
                    const int hh = col >> 6, dd = col & 63;
                    outH[((((size_t)bidx * NHEADS + hh) * seqlen + nn) << 6) + dd] =
                        (_Float16)v;
                } else {
                    outF[(size_t)row * N + col] = v + bias[col];
                }
            }
        }
    }
}

// ---------------------------------------------------------------------------
// Kernel 3: RoPE (in-place on Q/K head-layout; V roped + transposed to (bh,64,seq))
// rotate_half on dh=64: out[j] = x[j]*cos - x[j+32]*sin ; out[j+32] = x[j+32]*cos + x[j]*sin
// ---------------------------------------------------------------------------
__global__ void k_rope(_Float16* __restrict__ Qh, _Float16* __restrict__ Kh,
                       const _Float16* __restrict__ Vg, _Float16* __restrict__ Vt,
                       int seqlen) {
    const int idx = blockIdx.x * blockDim.x + threadIdx.x;
    const int total = BATCH * NHEADS * seqlen * 32;
    if (idx >= total) return;
    const int j = idx & 31;
    const int pos = (idx >> 5) % seqlen;
    const int bh = idx / (32 * seqlen);

    const float inv = __powf(10000.0f, -(float)j * (1.0f / 32.0f));
    const float ang = (float)pos * inv;
    float sn, cs;
    __sincosf(ang, &sn, &cs);

    const size_t base = ((size_t)bh * seqlen + pos) * DHEAD;
    {   // Q
        const float x1 = (float)Qh[base + j], x2 = (float)Qh[base + j + 32];
        Qh[base + j]      = (_Float16)(x1 * cs - x2 * sn);
        Qh[base + j + 32] = (_Float16)(x2 * cs + x1 * sn);
    }
    {   // K
        const float x1 = (float)Kh[base + j], x2 = (float)Kh[base + j + 32];
        Kh[base + j]      = (_Float16)(x1 * cs - x2 * sn);
        Kh[base + j + 32] = (_Float16)(x2 * cs + x1 * sn);
    }
    {   // V -> transposed (bh, d, pos)
        const float x1 = (float)Vg[base + j], x2 = (float)Vg[base + j + 32];
        const size_t tb = (size_t)bh * DHEAD * seqlen;
        Vt[tb + (size_t)j * seqlen + pos]        = (_Float16)(x1 * cs - x2 * sn);
        Vt[tb + (size_t)(j + 32) * seqlen + pos] = (_Float16)(x2 * cs + x1 * sn);
    }
}

// ---------------------------------------------------------------------------
// Kernel 4: causal flash attention (per (b,h), 64 query rows per block, 4 waves)
// Each wave: 16 query rows, online softmax over 64-key blocks.
// ---------------------------------------------------------------------------
__global__ __launch_bounds__(128)
void k_flash_attn(const _Float16* __restrict__ Qh, const _Float16* __restrict__ Kh,
                  const _Float16* __restrict__ Vt, const unsigned char* __restrict__ msk,
                  _Float16* __restrict__ AOh, int seqlen) {
    __shared__ _Float16 ldsP[4 * 16 * 64];   // per-wave 16x64 P tile (C->A relayout)

    const int lane = threadIdx.x & 31;
    const int wave = threadIdx.x >> 5;
    const int hv = lane >> 4, nl = lane & 15;
    const int h = blockIdx.y, b = blockIdx.z;
    const int q0 = blockIdx.x * 64 + wave * 16;
    const int bh = b * NHEADS + h;

    const _Float16* Qp = Qh + (size_t)bh * seqlen * DHEAD;
    const _Float16* Kp = Kh + (size_t)bh * seqlen * DHEAD;
    const _Float16* Vp = Vt + (size_t)bh * DHEAD * seqlen;
    const unsigned char* km = msk + (size_t)b * seqlen;
    _Float16* ldsW = ldsP + wave * (16 * 64);

    const float scale = 0.03125f;  // dim^-0.5 = 1024^-0.5 (reference scales by dim!)

    // Q fragments for the two k-steps over dh=64
    const v16h qa0 = load_a16(Qp, DHEAD, q0, 0, lane);
    const v16h qa1 = load_a16(Qp, DHEAD, q0, 32, lane);

    v8f o[4];
#pragma unroll
    for (int t = 0; t < 4; ++t) o[t] = v8f_zero();
    float mrow[8], lrow[8];
#pragma unroll
    for (int i = 0; i < 8; ++i) { mrow[i] = -1e30f; lrow[i] = 0.0f; }

    for (int j0 = 0; j0 < q0 + 16; j0 += 64) {   // causal: only blocks with keys <= q0+15
        // ---- S = Q K^T (16 x 64) ----
        v8f s[4];
#pragma unroll
        for (int t = 0; t < 4; ++t) {
            s[t] = v8f_zero();
            v16h kb0 = load_b16(Kp, DHEAD, j0 + 16 * t, 0, lane);
            s[t] = wmma_f16(qa0, kb0, s[t]);
            v16h kb1 = load_b16(Kp, DHEAD, j0 + 16 * t, 32, lane);
            s[t] = wmma_f16(qa1, kb1, s[t]);
        }
        // ---- scale + causal/input mask, row max ----
        float rmax[8];
#pragma unroll
        for (int i = 0; i < 8; ++i) rmax[i] = mrow[i];
#pragma unroll
        for (int t = 0; t < 4; ++t) {
#pragma unroll
            for (int i = 0; i < 8; ++i) {
                const int irow = q0 + i + 8 * hv;
                const int jcol = j0 + 16 * t + nl;
                float v = s[t][i] * scale;
                const bool ok = (jcol <= irow) && (km[jcol] != 0) && (km[irow] != 0);
                if (!ok) v = -1e30f;
                s[t][i] = v;
                rmax[i] = fmaxf(rmax[i], v);
            }
        }
#pragma unroll
        for (int i = 0; i < 8; ++i) {    // max over the 16 lanes of each half
            float r = rmax[i];
            r = fmaxf(r, __shfl_xor(r, 1, 32));
            r = fmaxf(r, __shfl_xor(r, 2, 32));
            r = fmaxf(r, __shfl_xor(r, 4, 32));
            r = fmaxf(r, __shfl_xor(r, 8, 32));
            rmax[i] = r;
        }
        // ---- online softmax update; write P tile to LDS as f16 ----
        float corr[8], rs[8];
#pragma unroll
        for (int i = 0; i < 8; ++i) {
            corr[i] = __expf(mrow[i] - rmax[i]);
            mrow[i] = rmax[i];
            rs[i] = 0.0f;
        }
#pragma unroll
        for (int t = 0; t < 4; ++t) {
#pragma unroll
            for (int i = 0; i < 8; ++i) {
                const float p = __expf(s[t][i] - mrow[i]);
                rs[i] += p;
                ldsW[(i + 8 * hv) * 64 + 16 * t + nl] = (_Float16)p;
            }
        }
#pragma unroll
        for (int i = 0; i < 8; ++i) {    // sum over the 16 lanes of each half
            float r = rs[i];
            r += __shfl_xor(r, 1, 32);
            r += __shfl_xor(r, 2, 32);
            r += __shfl_xor(r, 4, 32);
            r += __shfl_xor(r, 8, 32);
            lrow[i] = lrow[i] * corr[i] + r;
        }
#pragma unroll
        for (int t = 0; t < 4; ++t)
#pragma unroll
            for (int i = 0; i < 8; ++i) o[t][i] *= corr[i];

        // ---- O += P V : reload P from LDS in A layout ----
        wait_ds0();
        const v16h pa0 = load_a16(ldsW, 64, 0, 0, lane);
        const v16h pa1 = load_a16(ldsW, 64, 0, 32, lane);
#pragma unroll
        for (int t = 0; t < 4; ++t) {
            v16h vb0 = load_b16(Vp, seqlen, 16 * t, j0, lane);
            o[t] = wmma_f16(pa0, vb0, o[t]);
            v16h vb1 = load_b16(Vp, seqlen, 16 * t, j0 + 32, lane);
            o[t] = wmma_f16(pa1, vb1, o[t]);
        }
    }

    // ---- normalize and store (b, n, h*64) as f16 for the final GEMM ----
#pragma unroll
    for (int t = 0; t < 4; ++t) {
#pragma unroll
        for (int i = 0; i < 8; ++i) {
            const int irow = q0 + i + 8 * hv;
            const int col = h * DHEAD + 16 * t + nl;
            const float v = o[t][i] / fmaxf(lrow[i], 1e-20f);
            AOh[((size_t)b * seqlen + irow) * (NHEADS * DHEAD) + col] = (_Float16)v;
        }
    }
}

// ---------------------------------------------------------------------------
// Host launch
// ---------------------------------------------------------------------------
extern "C" void kernel_launch(void* const* d_in, const int* in_sizes, int n_in,
                              void* d_out, int out_size, void* d_ws, size_t ws_size,
                              hipStream_t stream) {
    (void)in_sizes; (void)n_in; (void)out_size; (void)ws_size;
    const float* x  = (const float*)d_in[0];
    const unsigned char* mask = (const unsigned char*)d_in[1];  // jnp bool = 1 byte
    const float* Wq = (const float*)d_in[2];
    const float* Wk = (const float*)d_in[3];
    const float* Wv = (const float*)d_in[4];
    const float* Wfc = (const float*)d_in[5];
    const float* bfc = (const float*)d_in[6];

    const int seqlen = SEQ, dm = DMODEL;
    const int M = BATCH * seqlen;                 // 4096
    const size_t xBytes = (size_t)M * dm * sizeof(_Float16);      // 8 MB
    const size_t wBytes = (size_t)dm * dm * sizeof(_Float16);     // 2 MB
    const size_t hBytes = (size_t)BATCH * NHEADS * seqlen * DHEAD * sizeof(_Float16); // 8 MB

    char* ws = (char*)d_ws;
    _Float16* xh  = (_Float16*)ws;            ws += xBytes;
    _Float16* wqh = (_Float16*)ws;            ws += wBytes;
    _Float16* wkh = (_Float16*)ws;            ws += wBytes;
    _Float16* wvh = (_Float16*)ws;            ws += wBytes;
    _Float16* wfh = (_Float16*)ws;            ws += wBytes;
    _Float16* Qh  = (_Float16*)ws;            ws += hBytes;
    _Float16* Kh  = (_Float16*)ws;            ws += hBytes;
    _Float16* Vg  = (_Float16*)ws;            ws += hBytes;
    _Float16* Vt  = (_Float16*)ws;            ws += hBytes;
    _Float16* AOh = (_Float16*)ws;            ws += hBytes;
    // total workspace: ~56 MB

    // 1) convert to f16
    {
        const int ne = M * dm;
        k_convert_f16<<<(ne + 255) / 256, 256, 0, stream>>>(x, xh, ne);
        const int nw = dm * dm;
        k_convert_f16<<<(nw + 255) / 256, 256, 0, stream>>>(Wq, wqh, nw);
        k_convert_f16<<<(nw + 255) / 256, 256, 0, stream>>>(Wk, wkh, nw);
        k_convert_f16<<<(nw + 255) / 256, 256, 0, stream>>>(Wv, wvh, nw);
        k_convert_f16<<<(nw + 255) / 256, 256, 0, stream>>>(Wfc, wfh, nw);
    }

    // 2) QKV projections -> f16 head layout  (block tile 64x256)
    dim3 gg(M / 64, dm / 256);
    k_gemm_wmma<<<gg, 256, 0, stream>>>(xh, wqh, M, dm, dm, seqlen, Qh, nullptr, nullptr, 0);
    k_gemm_wmma<<<gg, 256, 0, stream>>>(xh, wkh, M, dm, dm, seqlen, Kh, nullptr, nullptr, 0);
    k_gemm_wmma<<<gg, 256, 0, stream>>>(xh, wvh, M, dm, dm, seqlen, Vg, nullptr, nullptr, 0);

    // 3) RoPE on q,k (in place) and v (-> transposed Vt)
    {
        const int nt = BATCH * NHEADS * seqlen * 32;
        k_rope<<<(nt + 255) / 256, 256, 0, stream>>>(Qh, Kh, Vg, Vt, seqlen);
    }

    // 4) causal flash attention -> AOh (b, n, h*dh) f16
    {
        dim3 ga(seqlen / 64, NHEADS, BATCH);
        k_flash_attn<<<ga, 128, 0, stream>>>(Qh, Kh, Vt, mask, AOh, seqlen);
    }

    // 5) final projection + bias -> fp32 d_out
    k_gemm_wmma<<<gg, 256, 0, stream>>>(AOh, wfh, M, dm, dm, seqlen,
                                        nullptr, (float*)d_out, bfc, 1);
}
